// _MatrixDecomposition2DBase_69956427317520
// MI455X (gfx1250) — matrix operand order repaired and checked by reading, not compiled, not run on hardware
//
#include <hip/hip_runtime.h>

// ---------------- problem constants (from reference) ----------------
#define BB   16      // batches (MD_S == 1)
#define NN   512     // C  (N dimension)
#define DD   4096    // S  (D dimension)
#define RR   16      // MD_R
#define STEPS 7
#define INV_T 100.0f
#define EPS   1e-6f

typedef __attribute__((ext_vector_type(16))) __bf16 v16bf;
typedef __attribute__((ext_vector_type(8)))  float  v8f;

union U16 {
    v16bf  v;
    uint   u[8];
    uint4  q[2];
    ushort h[16];
};

// -------- bf16 helpers (bit-exact RNE, no reliance on __bf16 conversions) ----
__device__ inline ushort f2bf(float f) {
    uint u = __float_as_uint(f);
    u += 0x7FFFu + ((u >> 16) & 1u);
    return (ushort)(u >> 16);
}
__device__ inline float bf2f(ushort h) {
    return __uint_as_float(((uint)h) << 16);
}

// -------- WMMA wrapper: D = A(16x32 bf16) * B(32x16 bf16) + C(16x16 f32) ----
__device__ inline v8f wmma_bf16(v16bf a, v16bf b, v8f c) {
    return __builtin_amdgcn_wmma_f32_16x16x32_bf16(
        /*neg_a=*/false, a, /*neg_b=*/false, b,
        /*c_mod=*/(short)0, c, /*reuse_a=*/false, /*reuse_b=*/false);
}

// A-matrix 16x32 bf16 VGPR layout used here (contiguous K, not the ISA table order):
//   lanes 0-15: M=lane, lanes 16-31: M=lane-16
//   VGPR v holds K pair {2v,2v+1}; +16 for hi-lanes (contiguous K = 16*hi + e).
// Row-major source: A[m][k] = base[m*stride + k0 + K]  (pairs contiguous -> dword loads)
__device__ inline v16bf load_A_pairs(const ushort* base, int stride, int k0, int lane) {
    const int m  = lane & 15;
    const int hi = lane >> 4;
    U16 a;
    const ushort* p = base + (size_t)m * stride + k0;
#pragma unroll
    for (int vv = 0; vv < 8; ++vv) {
        const int bk = 2 * vv + hi * 16;
        a.u[vv] = *(const uint*)(p + bk);   // 2 bf16 per dword, K-contiguous
    }
    return a.v;
}

// Column-major source (X viewed as D x N): A[m][k] = base[K*stride] (base pre-offset by m)
__device__ inline v16bf load_A_strided(const ushort* base, int stride, int hi) {
    U16 a;
#pragma unroll
    for (int e = 0; e < 16; ++e) {
        const int vv = e >> 1, pos = e & 1;
        const int K = 2 * vv + hi * 16 + pos;
        a.h[e] = base[(size_t)K * stride];
    }
    return a.v;
}

// B-matrix 32x16 bf16: lane n = lane&15; lanes 0-15 hold K=0..15, lanes 16-31 K=16..31.
// Source mat is (Ncols x K) row-major (a transpose buffer): 16 contiguous elements.
__device__ inline v16bf load_B16(const ushort* mat, int stride, int k0, int lane) {
    const int r  = lane & 15;
    const int hi = lane >> 4;
    U16 b;
    const uint4* q = (const uint4*)(mat + (size_t)r * stride + k0 + hi * 16);
    b.q[0] = q[0];
    b.q[1] = q[1];
    return b.v;
}

// A = coef (N x R) zero-padded to K=32, from coefT (R x N): a.h[e] = coefT[K*NN + n]
__device__ inline v16bf load_A_pad16(const ushort* coefT, int n, int hi) {
    U16 a;
#pragma unroll
    for (int e = 0; e < 16; ++e) {
        const int vv = e >> 1, pos = e & 1;
        const int K = 2 * vv + hi * 16 + pos;
        a.h[e] = (K < RR) ? coefT[K * NN + n] : (ushort)0;
    }
    return a.v;
}

// B = bases (R x 16d cols) zero-padded to K=32, from basesT (R x D)
__device__ inline v16bf load_B_pad16(const ushort* basesT, int d, int hi) {
    U16 b;
#pragma unroll
    for (int e = 0; e < 16; ++e)
        b.h[e] = hi ? (ushort)0 : basesT[e * DD + d];
    return b.v;
}

// num(N x R) tile for this wave: Xrow(N x D) @ basesT^T, K over D in chunks of 32
__device__ inline v8f gemm_xtb(const ushort* Xb, const ushort* basesT, int n0, int lane) {
    v8f acc = {};
    const ushort* xrow = Xb + (size_t)n0 * DD;
    for (int d0 = 0; d0 < DD; d0 += 32) {
        __builtin_prefetch(xrow + d0 + 64, 0, 1);          // global_prefetch_b8
        v16bf a = load_A_pairs(xrow, DD, d0, lane);        // X row-major tile
        v16bf b = load_B16(basesT, DD, d0, lane);          // basesT rows (LDS)
        acc = wmma_bf16(a, b, acc);
    }
    return acc;
}

// Gram G(16x16) = mat * mat^T over K=klen, mat is (16 x klen) row-major in LDS.
// Partial per-wave accumulators merged with shared atomicAdd (ds_add_f32).
__device__ inline void compute_gram(const ushort* mat, int klen, float* G,
                                    int tid, int lane, int wave, int col, int hi) {
    if (tid < 256) G[tid] = 0.f;
    __syncthreads();
    v8f acc = {};
    const int chunks = klen / 32;
    for (int ch = wave; ch < chunks; ch += 32) {
        const int k0 = ch * 32;
        v16bf a = load_A_pairs(mat, klen, k0, lane);
        v16bf b = load_B16(mat, klen, k0, lane);
        acc = wmma_bf16(a, b, acc);
    }
#pragma unroll
    for (int v = 0; v < 8; ++v) {
        const int row = v + hi * 8;
        atomicAdd(&G[row * 16 + col], acc[v]);
    }
    __syncthreads();
}

// coef <- coef * num / (coef @ BtB + eps); wave owns n-tile n0..n0+15
__device__ inline void coef_update(v8f num, float* coefF, ushort* coefT,
                                   const float* BtB, int n0, int col, int hi) {
    float cn[8];
#pragma unroll
    for (int v = 0; v < 8; ++v) {
        const int n = n0 + v + hi * 8;
        float den = 0.f;
#pragma unroll
        for (int s = 0; s < RR; ++s)
            den += coefF[s * NN + n] * BtB[s * 16 + col];
        cn[v] = coefF[col * NN + n] * num[v] / (den + EPS);
    }
    __syncthreads();                 // all old-coef reads done before any writes
#pragma unroll
    for (int v = 0; v < 8; ++v) {
        const int n = n0 + v + hi * 8;
        coefF[col * NN + n] = cn[v];
        coefT[col * NN + n] = f2bf(cn[v]);
    }
    __syncthreads();
}

// ---------------------------------------------------------------------------
__global__ void cvt_bf16_kernel(const float* __restrict__ x,
                                ushort* __restrict__ xb, long n) {
    long i = (long)blockIdx.x * blockDim.x + threadIdx.x;
    const long stride = (long)gridDim.x * blockDim.x;
    for (; i < n; i += stride) xb[i] = f2bf(x[i]);
}

// One workgroup per batch. 1024 threads = 32 wave32 waves.
// LDS (dynamic): basesT bf16 [R][D] 128KB | coefF f32 [R][N] 32KB |
//                coefT bf16 [R][N] 16KB  | BtB 1KB | CtC 1KB
__global__ __launch_bounds__(1024, 1)
void nmf_fused_kernel(const ushort* __restrict__ xall,
                      const float* __restrict__ bases_in,
                      float* __restrict__ out) {
    extern __shared__ char smem[];
    ushort* basesT = (ushort*)smem;                                   // [RR][DD]
    float*  coefF  = (float*)(smem + 2 * RR * DD);                    // [RR][NN]
    ushort* coefT  = (ushort*)(smem + 2 * RR * DD + 4 * RR * NN);     // [RR][NN]
    float*  BtB    = (float*)(smem + 2 * RR * DD + 4 * RR * NN + 2 * RR * NN);
    float*  CtC    = BtB + 256;

    const int b    = blockIdx.x;
    const int tid  = threadIdx.x;
    const int lane = tid & 31;
    const int wave = tid >> 5;
    const int col  = lane & 15;
    const int hi   = lane >> 4;
    const int n0   = wave * 16;           // this wave's coef n-tile
    const ushort* Xb = xall + (size_t)b * NN * DD;

    // ---- load bases (B, D, R) f32 -> basesT (R, D) bf16 in LDS ----
    const float* Bin = bases_in + (size_t)b * DD * RR;
    for (int i = tid; i < DD * RR; i += 1024) {
        const int d = i >> 4, r = i & 15;
        basesT[r * DD + d] = f2bf(Bin[i]);
    }
    __syncthreads();

    // ---- coef init: softmax_r(inv_t * X^T bases) ----
    {
        v8f num = gemm_xtb(Xb, basesT, n0, lane);
#pragma unroll
        for (int v = 0; v < 8; ++v) {
            float t = INV_T * num[v];
            float mx = t;
#pragma unroll
            for (int off = 1; off < 16; off <<= 1)
                mx = fmaxf(mx, __shfl_xor(mx, off, 32));
            float e = __expf(t - mx);
            float sm = e;
#pragma unroll
            for (int off = 1; off < 16; off <<= 1)
                sm += __shfl_xor(sm, off, 32);
            const float c = e / sm;
            const int n = n0 + v + hi * 8;
            coefF[col * NN + n] = c;
            coefT[col * NN + n] = f2bf(c);
        }
        __syncthreads();
    }

    // ---- 7 NMF multiplicative-update steps + final compute_coef ----
    for (int it = 0; it <= STEPS; ++it) {
        // coef update: num = X^T bases ; den = coef (bases^T bases)
        v8f num = gemm_xtb(Xb, basesT, n0, lane);
        compute_gram(basesT, DD, BtB, tid, lane, wave, col, hi);
        coef_update(num, coefF, coefT, BtB, n0, col, hi);

        if (it == STEPS) break;       // compute_coef: coef updated, bases kept

        // bases update: num2 = X coef ; den2 = bases (coef^T coef)
        compute_gram(coefT, NN, CtC, tid, lane, wave, col, hi);
        for (int t = wave; t < DD / 16; t += 32) {
            const int d0 = t * 16;
            v8f acc = {};
            for (int ch = 0; ch < NN / 32; ++ch) {
                const int nk = ch * 32;
                v16bf a = load_A_strided(Xb + (size_t)nk * DD + d0 + col, DD, hi);
                v16bf bm = load_B16(coefT, NN, nk, lane);
                acc = wmma_bf16(a, bm, acc);
            }
            float bn[8];
#pragma unroll
            for (int v = 0; v < 8; ++v) {
                const int d = d0 + v + hi * 8;
                float den = 0.f;
#pragma unroll
                for (int s = 0; s < RR; ++s)
                    den += bf2f(basesT[s * DD + d]) * CtC[s * 16 + col];
                bn[v] = bf2f(basesT[col * DD + d]) * acc[v] / (den + EPS);
            }
#pragma unroll
            for (int v = 0; v < 8; ++v) {        // columns disjoint per tile:
                const int d = d0 + v + hi * 8;   // wave-local in-order LDS, no race
                basesT[col * DD + d] = f2bf(bn[v]);
            }
        }
        __syncthreads();
    }

    // ---- out(n,d) = sum_r coef[n][r] * bases[d][r]  (K=16 padded to 32) ----
    float* outb = out + (size_t)b * NN * DD;
    for (int t = wave; t < (NN / 16) * (DD / 16); t += 32) {
        const int nt = t / (DD / 16);
        const int dt = t % (DD / 16);
        const int on0 = nt * 16, od0 = dt * 16;
        v16bf a = load_A_pad16(coefT, on0 + col, hi);
        v16bf bm = load_B_pad16(basesT, od0 + col, hi);
        v8f acc = {};
        acc = wmma_bf16(a, bm, acc);
#pragma unroll
        for (int v = 0; v < 8; ++v) {
            const int n = on0 + v + hi * 8;
            outb[(size_t)n * DD + od0 + col] = acc[v];   // coalesced over lanes
        }
    }
}

extern "C" void kernel_launch(void* const* d_in, const int* in_sizes, int n_in,
                              void* d_out, int out_size, void* d_ws, size_t ws_size,
                              hipStream_t stream) {
    const float* x     = (const float*)d_in[0];   // (B, C, S) f32
    const float* bases = (const float*)d_in[1];   // (B, S, R) f32 (normalized)
    float* out = (float*)d_out;                   // (B, C, S) f32
    ushort* xb = (ushort*)d_ws;                   // bf16 copy of x: 64 MiB scratch

    const long n = (long)BB * NN * DD;
    cvt_bf16_kernel<<<4096, 256, 0, stream>>>(x, xb, n);

    const size_t sh = 2 * RR * DD      // basesT bf16
                    + 4 * RR * NN      // coefF f32
                    + 2 * RR * NN      // coefT bf16
                    + 2 * 256 * 4;     // BtB + CtC
    nmf_fused_kernel<<<BB, 1024, sh, stream>>>(xb, bases, out);
}
